// AttentionBlock_80341658239008
// MI455X (gfx1250) — matrix operand order repaired and checked by reading, not compiled, not run on hardware
//
#include <hip/hip_runtime.h>
#include <hip/hip_bf16.h>

// ---------------------------------------------------------------------------
// AttentionBlock on MI455X (gfx1250), bf16 WMMA pipeline, register-macrotiled.
//   Each GEMM wave computes a 32x64 tile: 2 A-frags x 4 B-frags -> 8 WMMAs per
//   32-deep k-step (6 fragment loads / 8 matrix ops). Per-batch S (f32) and
//   P (bf16) buffers are reused so attention traffic stays in the 192MB L2.
// ---------------------------------------------------------------------------

typedef __bf16 bf16_t;
typedef __attribute__((ext_vector_type(16))) __bf16 v16bf;
typedef __attribute__((ext_vector_type(8)))  __bf16 v8bf;
typedef __attribute__((ext_vector_type(8)))  float  v8f;

#define CC     512
#define NN     4096          // H*W
#define BB     4
#define ROWS   (BB * NN)     // 16384
#define SCALE  0.044194173824159216f   // 512^-0.5

// ---- workspace layout (bytes) ---------------------------------------------
static constexpr size_t OFF_STATS = 0;                              // 128*2 f32
static constexpr size_t OFF_H     = 4096;                           // bf16 [row,c]
static constexpr size_t OFF_WQ    = OFF_H  + (size_t)ROWS * CC * 2;
static constexpr size_t OFF_WK    = OFF_WQ + (size_t)CC * CC * 2;
static constexpr size_t OFF_WV    = OFF_WK + (size_t)CC * CC * 2;
static constexpr size_t OFF_WO    = OFF_WV + (size_t)CC * CC * 2;
static constexpr size_t OFF_Q     = OFF_WO + (size_t)CC * CC * 2;   // [row,c]
static constexpr size_t OFF_K     = OFF_Q  + (size_t)ROWS * CC * 2; // [row,c]
static constexpr size_t OFF_VT    = OFF_K  + (size_t)ROWS * CC * 2; // [b][c][n]
static constexpr size_t OFF_O     = OFF_VT + (size_t)ROWS * CC * 2; // [row,c]
static constexpr size_t OFF_S     = OFF_O  + (size_t)ROWS * CC * 2; // [n][m] f32
static constexpr size_t OFF_P     = OFF_S  + (size_t)NN * NN * 4;   // [n][m] bf16

// ---- WMMA helpers ----------------------------------------------------------
__device__ __forceinline__ v8f wmma_bf16(v16bf a, v16bf b, v8f c) {
  return __builtin_amdgcn_wmma_f32_16x16x32_bf16(
      false, a, false, b, (short)0, c, false, false);
}

// A fragment 16x32 bf16 from row-major [M,K]; p at (row0,k0).
__device__ __forceinline__ v16bf load_a_frag(const bf16_t* p, int stride, int lane) {
  const bf16_t* q = p + (size_t)(lane & 15) * stride + ((lane >> 4) << 4);
  return *(const v16bf*)q;
}

// B fragment 32x16 bf16 from row-major [N,K] source.
__device__ __forceinline__ v16bf load_b_frag(const bf16_t* p, int stride, int lane) {
  const bf16_t* q = p + (size_t)(lane & 15) * stride + ((lane >> 4) << 4);
  return *(const v16bf*)q;
}

// One 32-deep k-step of the 2x4 macro-tile: 6 fragment loads, 8 WMMAs.
__device__ __forceinline__ void mma_step_2x4(const bf16_t* a0p, const bf16_t* a1p,
                                             const bf16_t* b0p, const bf16_t* b1p,
                                             const bf16_t* b2p, const bf16_t* b3p,
                                             int as, int bs, int lane, v8f* acc) {
  v16bf a0 = load_a_frag(a0p, as, lane);
  v16bf a1 = load_a_frag(a1p, as, lane);
  v16bf b0 = load_b_frag(b0p, bs, lane);
  v16bf b1 = load_b_frag(b1p, bs, lane);
  v16bf b2 = load_b_frag(b2p, bs, lane);
  v16bf b3 = load_b_frag(b3p, bs, lane);
  acc[0] = wmma_bf16(a0, b0, acc[0]);
  acc[1] = wmma_bf16(a0, b1, acc[1]);
  acc[2] = wmma_bf16(a0, b2, acc[2]);
  acc[3] = wmma_bf16(a0, b3, acc[3]);
  acc[4] = wmma_bf16(a1, b0, acc[4]);
  acc[5] = wmma_bf16(a1, b1, acc[5]);
  acc[6] = wmma_bf16(a1, b2, acc[6]);
  acc[7] = wmma_bf16(a1, b3, acc[7]);
}

// ---- GroupNorm stats: one block per (b, group) -----------------------------
__global__ void gn_stats_kernel(const float* __restrict__ x, float* __restrict__ stats) {
  int bg = blockIdx.x;                    // 0..127
  int b = bg >> 5, g = bg & 31;
  const float* base = x + ((size_t)b * CC + (size_t)g * 16) * NN;
  float s = 0.f, sq = 0.f;
  for (int i = threadIdx.x; i < 16 * NN; i += 256) {
    float v = base[i];
    s += v; sq += v * v;
  }
  __shared__ float ss[256], ssq[256];
  ss[threadIdx.x] = s; ssq[threadIdx.x] = sq;
  __syncthreads();
  for (int w = 128; w > 0; w >>= 1) {
    if (threadIdx.x < w) { ss[threadIdx.x] += ss[threadIdx.x + w]; ssq[threadIdx.x] += ssq[threadIdx.x + w]; }
    __syncthreads();
  }
  if (threadIdx.x == 0) {
    float mean = ss[0] * (1.0f / 65536.0f);
    float var  = ssq[0] * (1.0f / 65536.0f) - mean * mean;
    stats[bg * 2 + 0] = mean;
    stats[bg * 2 + 1] = rsqrtf(var + 1e-5f);
  }
}

// ---- GroupNorm apply + transpose [b,c,n] -> [row,c] bf16 -------------------
__global__ void gn_apply_kernel(const float* __restrict__ x, const float* __restrict__ gw,
                                const float* __restrict__ gb, const float* __restrict__ stats,
                                bf16_t* __restrict__ h) {
  size_t idx = (size_t)blockIdx.x * 256 + threadIdx.x;
  int c = (int)(idx & (CC - 1));
  size_t r = idx >> 9;
  int b = (int)(r >> 12);
  int n = (int)(r & (NN - 1));
  int g = c >> 4;
  float mean = stats[(b * 32 + g) * 2 + 0];
  float rstd = stats[(b * 32 + g) * 2 + 1];
  float v = x[((size_t)b * CC + c) * NN + n];
  h[idx] = (bf16_t)(((v - mean) * rstd) * gw[c] + gb[c]);
}

// ---- Convert the four [C,C] weights f32 -> bf16 ----------------------------
__global__ void convert_w_kernel(const float* __restrict__ wq, const float* __restrict__ wk,
                                 const float* __restrict__ wv, const float* __restrict__ wo,
                                 bf16_t* __restrict__ dst) {
  size_t idx = (size_t)blockIdx.x * 256 + threadIdx.x;   // 4 * 262144
  int sel = (int)(idx >> 18);
  int off = (int)(idx & 262143);
  const float* src = (sel == 0) ? wq : (sel == 1) ? wk : (sel == 2) ? wv : wo;
  dst[idx] = (bf16_t)src[off];
}

// ---- Projection GEMM: D = A(16384x512) @ W^T(512x512) + bias ---------------
// trans==0: D row-major [row, o] bf16;  trans==1: D = VT [b][o][n] bf16.
__global__ void proj_gemm_kernel(const bf16_t* __restrict__ A, const bf16_t* __restrict__ W,
                                 const float* __restrict__ bias, bf16_t* __restrict__ D,
                                 int trans) {
  int lane = threadIdx.x & 31;
  int wave = threadIdx.x >> 5;
  int n0 = blockIdx.x * 64;                      // output-channel macro tile
  int m0 = (blockIdx.y * 8 + wave) * 32;         // token-row macro tile
  const bf16_t* a0 = A + (size_t)m0 * CC;
  const bf16_t* a1 = a0 + (size_t)16 * CC;
  const bf16_t* w0 = W + (size_t)n0 * CC;
  v8f acc[8] = {};
#pragma unroll 2
  for (int kc = 0; kc < CC; kc += 32) {
    __builtin_prefetch(a0 + kc + 512, 0, 3);
    __builtin_prefetch(a1 + kc + 512, 0, 3);
    mma_step_2x4(a0 + kc, a1 + kc,
                 w0 + kc, w0 + (size_t)16 * CC + kc,
                 w0 + (size_t)32 * CC + kc, w0 + (size_t)48 * CC + kc,
                 CC, CC, lane, acc);
  }
  int col = lane & 15, rb = (lane >> 4) << 3;
#pragma unroll
  for (int j = 0; j < 4; ++j) {
    int o = n0 + j * 16 + col;
    float bb = bias[o];
#pragma unroll
    for (int i = 0; i < 2; ++i) {
      v8f a = acc[i * 4 + j];
      int mb = m0 + i * 16 + rb;
#pragma unroll
      for (int r = 0; r < 8; ++r) {
        int row = mb + r;
        if (trans) {
          int b = row >> 12, n = row & (NN - 1);
          D[((size_t)b * CC + o) * NN + n] = (bf16_t)(a[r] + bb);
        } else {
          D[(size_t)row * CC + o] = (bf16_t)(a[r] + bb);
        }
      }
    }
  }
}

// ---- S = scale * Q K^T for one batch (f32 out, lives in L2) ----------------
__global__ void attn_scores_kernel(const bf16_t* __restrict__ Q, const bf16_t* __restrict__ K,
                                   float* __restrict__ S, int b) {
  int lane = threadIdx.x & 31;
  int wave = threadIdx.x >> 5;
  int k0 = blockIdx.x * 64;                      // key macro tile
  int q0 = (blockIdx.y * 8 + wave) * 32;         // query macro tile
  const bf16_t* qb = Q + ((size_t)b * NN + q0) * CC;
  const bf16_t* kb = K + ((size_t)b * NN + k0) * CC;
  v8f acc[8] = {};
#pragma unroll 2
  for (int kc = 0; kc < CC; kc += 32) {
    __builtin_prefetch(qb + kc + 512, 0, 3);
    __builtin_prefetch(qb + (size_t)16 * CC + kc + 512, 0, 3);
    mma_step_2x4(qb + kc, qb + (size_t)16 * CC + kc,
                 kb + kc, kb + (size_t)16 * CC + kc,
                 kb + (size_t)32 * CC + kc, kb + (size_t)48 * CC + kc,
                 CC, CC, lane, acc);
  }
  int col = lane & 15, rb = (lane >> 4) << 3;
#pragma unroll
  for (int j = 0; j < 4; ++j)
#pragma unroll
    for (int i = 0; i < 2; ++i) {
      v8f a = acc[i * 4 + j];
#pragma unroll
      for (int r = 0; r < 8; ++r)
        S[(size_t)(q0 + i * 16 + rb + r) * NN + k0 + j * 16 + col] = a[r] * SCALE;
    }
}

// ---- Row softmax: S (f32) -> P (bf16); one block per row -------------------
__global__ void softmax_kernel(const float* __restrict__ S, bf16_t* __restrict__ P) {
  int row = blockIdx.x;
  const float* s = S + (size_t)row * NN;
  bf16_t* p = P + (size_t)row * NN;
  __shared__ float red[256];
  float m = -3.4e38f;
  for (int i = threadIdx.x; i < NN; i += 256) m = fmaxf(m, s[i]);
  red[threadIdx.x] = m; __syncthreads();
  for (int w = 128; w > 0; w >>= 1) {
    if (threadIdx.x < w) red[threadIdx.x] = fmaxf(red[threadIdx.x], red[threadIdx.x + w]);
    __syncthreads();
  }
  m = red[0]; __syncthreads();
  float sum = 0.f;
  for (int i = threadIdx.x; i < NN; i += 256) sum += __expf(s[i] - m);
  red[threadIdx.x] = sum; __syncthreads();
  for (int w = 128; w > 0; w >>= 1) {
    if (threadIdx.x < w) red[threadIdx.x] += red[threadIdx.x + w];
    __syncthreads();
  }
  float inv = 1.0f / red[0];
  for (int i = threadIdx.x; i < NN; i += 256) p[i] = (bf16_t)(__expf(s[i] - m) * inv);
}

// ---- O = P V for one batch (V pre-transposed [c][n]) -----------------------
__global__ void attn_v_kernel(const bf16_t* __restrict__ P, const bf16_t* __restrict__ VT,
                              bf16_t* __restrict__ O, int b) {
  int lane = threadIdx.x & 31;
  int wave = threadIdx.x >> 5;
  int c0 = blockIdx.x * 64;                      // channel macro tile
  int q0 = (blockIdx.y * 8 + wave) * 32;         // query macro tile
  const bf16_t* pb = P + (size_t)q0 * NN;
  const bf16_t* vb = VT + (size_t)b * CC * NN + (size_t)c0 * NN;
  v8f acc[8] = {};
#pragma unroll 2
  for (int kc = 0; kc < NN; kc += 32) {          // 128 k-steps over keys
    __builtin_prefetch(pb + kc + 512, 0, 3);
    __builtin_prefetch(pb + (size_t)16 * NN + kc + 512, 0, 3);
    mma_step_2x4(pb + kc, pb + (size_t)16 * NN + kc,
                 vb + kc, vb + (size_t)16 * NN + kc,
                 vb + (size_t)32 * NN + kc, vb + (size_t)48 * NN + kc,
                 NN, NN, lane, acc);
  }
  int col = lane & 15, rb = (lane >> 4) << 3;
  bf16_t* ob = O + (size_t)b * NN * CC;
#pragma unroll
  for (int j = 0; j < 4; ++j)
#pragma unroll
    for (int i = 0; i < 2; ++i) {
      v8f a = acc[i * 4 + j];
#pragma unroll
      for (int r = 0; r < 8; ++r)
        ob[(size_t)(q0 + i * 16 + rb + r) * CC + c0 + j * 16 + col] = (bf16_t)a[r];
    }
}

// ---- out = x + O @ Wo^T + bo, written [b][o][n] f32 ------------------------
__global__ void out_proj_kernel(const bf16_t* __restrict__ O, const bf16_t* __restrict__ wo,
                                const float* __restrict__ bo, const float* __restrict__ x,
                                float* __restrict__ out) {
  int lane = threadIdx.x & 31;
  int wave = threadIdx.x >> 5;
  int o0 = blockIdx.x * 64;
  int m0 = (blockIdx.y * 8 + wave) * 32;
  const bf16_t* a0 = O + (size_t)m0 * CC;
  const bf16_t* a1 = a0 + (size_t)16 * CC;
  const bf16_t* w0 = wo + (size_t)o0 * CC;
  v8f acc[8] = {};
#pragma unroll 2
  for (int kc = 0; kc < CC; kc += 32) {
    __builtin_prefetch(a0 + kc + 512, 0, 3);
    __builtin_prefetch(a1 + kc + 512, 0, 3);
    mma_step_2x4(a0 + kc, a1 + kc,
                 w0 + kc, w0 + (size_t)16 * CC + kc,
                 w0 + (size_t)32 * CC + kc, w0 + (size_t)48 * CC + kc,
                 CC, CC, lane, acc);
  }
  int col = lane & 15, rb = (lane >> 4) << 3;
  int b = m0 >> 12, n = m0 & (NN - 1);
#pragma unroll
  for (int j = 0; j < 4; ++j) {
    int o = o0 + j * 16 + col;
    float bb = bo[o];
#pragma unroll
    for (int i = 0; i < 2; ++i) {
      v8f a = acc[i * 4 + j];
#pragma unroll
      for (int r = 0; r < 8; ++r) {
        size_t xi = ((size_t)b * CC + o) * NN + (n + i * 16 + rb + r);
        out[xi] = x[xi] + a[r] + bb;
      }
    }
  }
}

// ---------------------------------------------------------------------------
extern "C" void kernel_launch(void* const* d_in, const int* in_sizes, int n_in,
                              void* d_out, int out_size, void* d_ws, size_t ws_size,
                              hipStream_t stream) {
  const float* x  = (const float*)d_in[0];
  const float* gw = (const float*)d_in[1];
  const float* gb = (const float*)d_in[2];
  const float* wq = (const float*)d_in[3];
  const float* bq = (const float*)d_in[4];
  const float* wk = (const float*)d_in[5];
  const float* bk = (const float*)d_in[6];
  const float* wv = (const float*)d_in[7];
  const float* bv = (const float*)d_in[8];
  const float* wo = (const float*)d_in[9];
  const float* bo = (const float*)d_in[10];
  float* out = (float*)d_out;

  char* ws = (char*)d_ws;
  float*  stats = (float*)(ws + OFF_STATS);
  bf16_t* H  = (bf16_t*)(ws + OFF_H);
  bf16_t* WQ = (bf16_t*)(ws + OFF_WQ);
  bf16_t* WK = (bf16_t*)(ws + OFF_WK);
  bf16_t* WV = (bf16_t*)(ws + OFF_WV);
  bf16_t* WO = (bf16_t*)(ws + OFF_WO);
  bf16_t* Q  = (bf16_t*)(ws + OFF_Q);
  bf16_t* K  = (bf16_t*)(ws + OFF_K);
  bf16_t* VT = (bf16_t*)(ws + OFF_VT);
  bf16_t* O  = (bf16_t*)(ws + OFF_O);
  float*  S  = (float*)(ws + OFF_S);
  bf16_t* P  = (bf16_t*)(ws + OFF_P);

  // 1. GroupNorm
  gn_stats_kernel<<<128, 256, 0, stream>>>(x, stats);
  gn_apply_kernel<<<(ROWS * CC) / 256, 256, 0, stream>>>(x, gw, gb, stats, H);
  // 2. Weights to bf16
  convert_w_kernel<<<(4 * CC * CC) / 256, 256, 0, stream>>>(wq, wk, wv, wo, WQ);
  // 3. Q / K / V^T projections (H stays L2-resident across the three passes)
  dim3 pgrid(CC / 64, ROWS / 32 / 8);
  proj_gemm_kernel<<<pgrid, 256, 0, stream>>>(H, WQ, bq, Q, 0);
  proj_gemm_kernel<<<pgrid, 256, 0, stream>>>(H, WK, bk, K, 0);
  proj_gemm_kernel<<<pgrid, 256, 0, stream>>>(H, WV, bv, VT, 1);
  // 4. Attention per batch; S/P buffers reused so they stay resident in L2.
  for (int b = 0; b < BB; ++b) {
    attn_scores_kernel<<<dim3(NN / 64, NN / 32 / 8), 256, 0, stream>>>(Q, K, S, b);
    softmax_kernel<<<NN, 256, 0, stream>>>(S, P);
    attn_v_kernel<<<dim3(CC / 64, NN / 32 / 8), 256, 0, stream>>>(P, VT, O, b);
  }
  // 5. Output projection + residual
  out_proj_kernel<<<dim3(CC / 64, ROWS / 32 / 8), 256, 0, stream>>>(O, WO, bo, x, out);
}
